// CausalSelfAttention_27135603376642
// MI455X (gfx1250) — compile-verified, matrix operand order checked by reading, not run on hardware
//
#include <hip/hip_runtime.h>

// ---------------- problem constants (from reference) ----------------
#define B_   4
#define T_   2048
#define C_   1024          // D_MODEL
#define H_   16
#define DH   64
#define BT_  (B_ * T_)     // 8192

#if __has_builtin(__builtin_amdgcn_tensor_load_to_lds)
#define HAS_TDM 1
#else
#define HAS_TDM 0
#endif

// ---------------- CDNA5 WMMA types ----------------
typedef __attribute__((ext_vector_type(16))) __bf16 v16bf;
typedef __attribute__((ext_vector_type(8)))  __bf16 v8bf;
typedef __attribute__((ext_vector_type(8)))  float  v8f;
typedef __attribute__((ext_vector_type(4)))  unsigned int u32x4;
typedef __attribute__((ext_vector_type(8)))  int i32x8;
typedef __attribute__((ext_vector_type(4)))  int i32x4;

__device__ __forceinline__ v8f wmma_bf16(v16bf a, v16bf b, v8f c) {
  return __builtin_amdgcn_wmma_f32_16x16x32_bf16(
      false, a, false, b, (short)0, c, false, false);
}

__device__ __forceinline__ unsigned short f2bf(float f) {
  union { float f; unsigned u; } x; x.f = f;
  unsigned r = x.u + (0x7FFFu + ((x.u >> 16) & 1u));
  return (unsigned short)(r >> 16);
}

// 16x32 bf16 operand tile load (A layout; B identical when operand stored
// with its N dimension as rows). p -> element [row][kbase + khalf].
__device__ __forceinline__ v16bf load_tile16x32(const unsigned short* p) {
  v8bf lo = *(const v8bf*)(p);
  v8bf hi = *(const v8bf*)(p + 16);
  return __builtin_shufflevector(lo, hi, 0,1,2,3,4,5,6,7,8,9,10,11,12,13,14,15);
}

#if HAS_TDM
// 2D TENSOR_LOAD_TO_LDS: tile_d1 rows of tile_d0 bf16 elements, row stride
// stride0 (elements), packed contiguously into LDS at lds_byte_off.
__device__ __forceinline__ void tdm_load_2d(const unsigned short* gptr,
                                            unsigned lds_byte_off,
                                            unsigned tile_d0, unsigned tile_d1,
                                            unsigned tensor_d0, unsigned tensor_d1,
                                            unsigned stride0) {
  unsigned long long ga = (unsigned long long)gptr;
  unsigned ga_lo = (unsigned)__builtin_amdgcn_readfirstlane((int)(unsigned)ga);
  unsigned ga_hi = (unsigned)__builtin_amdgcn_readfirstlane((int)(unsigned)(ga >> 32));
  unsigned lds   = (unsigned)__builtin_amdgcn_readfirstlane((int)lds_byte_off);
  u32x4 g0;
  g0[0] = 1u;                                        // count=1 (valid, user mode)
  g0[1] = lds;                                       // lds_addr
  g0[2] = ga_lo;                                     // global_addr[31:0]
  g0[3] = (ga_hi & 0x01FFFFFFu) | 0x80000000u;       // global_addr[56:32] | type=2
  i32x8 g1;
  g1[0] = 0x00010000;                                // data_size=1 (2 bytes)
  g1[1] = (int)(tensor_d0 << 16);                    // abar=0 | tensor_dim0 lo16
  g1[2] = (int)((tensor_d0 >> 16) | (tensor_d1 << 16));
  g1[3] = (int)((tensor_d1 >> 16) | (tile_d0 << 16));
  g1[4] = (int)(tile_d1 & 0xFFFFu);                  // tile_dim1 | tile_dim2=0
  g1[5] = (int)stride0;                              // tensor_dim0_stride lo32
  g1[6] = 0;                                         // stride hi | dim1_stride lo
  g1[7] = 0;
  i32x4 z = {0, 0, 0, 0};
#if __clang_major__ >= 23
  i32x8 z8 = {0, 0, 0, 0, 0, 0, 0, 0};
  __builtin_amdgcn_tensor_load_to_lds(g0, g1, z, z, z8, 0);
#else
  __builtin_amdgcn_tensor_load_to_lds(g0, g1, z, z, 0);
#endif
}
#endif  // HAS_TDM

// ---------------- pre-pass kernels ----------------
__global__ __launch_bounds__(256) void cvt_f32_bf16(const float* __restrict__ src,
                                                    unsigned short* __restrict__ dst,
                                                    int n) {
  int i = blockIdx.x * 256 + threadIdx.x;
  if (i < n) dst[i] = f2bf(src[i]);
}

// dst[n][k] = bf16(src[k][n]); src [ROWS][COLS] f32 -> dst [COLS][ROWS] bf16
template <int ROWS, int COLS>
__global__ __launch_bounds__(256) void transpose_cvt(const float* __restrict__ src,
                                                     unsigned short* __restrict__ dst) {
  int i = blockIdx.x * 256 + threadIdx.x;
  if (i < ROWS * COLS) {
    int kq = i & (ROWS - 1);
    int n  = i / ROWS;
    dst[i] = f2bf(src[kq * COLS + n]);
  }
}

// ---------------- kernel 1: QKV projection GEMM (32x64 per wave) ----------------
__global__ __launch_bounds__(256) void gemm_qkv(const unsigned short* __restrict__ xb,
                                                const unsigned short* __restrict__ wT,
                                                unsigned short* __restrict__ q,
                                                unsigned short* __restrict__ k,
                                                unsigned short* __restrict__ vt) {
  const int lane = threadIdx.x & 31;
  const int wave = threadIdx.x >> 5;
  const int w    = blockIdx.x * 8 + wave;
  const int tm   = w / 48;                 // 256 macro rows (32 tokens)
  const int tn   = w % 48;                 // 48 macro cols (64 outputs)
  const int l15  = lane & 15;
  const int kh   = (lane >> 4) * 8;

  const unsigned short* pa0 = xb + (size_t)(tm * 32 + l15) * C_ + kh;
  const unsigned short* pa1 = pa0 + 16 * C_;
  const unsigned short* pb  = wT + (size_t)(tn * 64 + l15) * C_ + kh;

  v8f acc[2][4] = {};
  for (int kk = 0; kk < C_; kk += 32) {
    __builtin_prefetch(pa0 + ((kk + 256) & (C_ - 1)), 0, 3);  // wraps in-row: safe
    __builtin_prefetch(pb  + ((kk + 256) & (C_ - 1)), 0, 3);
    v16bf a0 = load_tile16x32(pa0 + kk);
    v16bf a1 = load_tile16x32(pa1 + kk);
    #pragma unroll
    for (int j = 0; j < 4; ++j) {
      v16bf b = load_tile16x32(pb + (size_t)j * 16 * C_ + kk);
      acc[0][j] = wmma_bf16(a0, b, acc[0][j]);
      acc[1][j] = wmma_bf16(a1, b, acc[1][j]);
    }
  }

  #pragma unroll
  for (int i = 0; i < 2; ++i) {
    #pragma unroll
    for (int j = 0; j < 4; ++j) {
      const int nn    = tn * 64 + j * 16 + l15;
      const int which = nn >> 10;
      const int hh    = (nn >> 6) & (H_ - 1);
      const int d     = nn & (DH - 1);
      #pragma unroll
      for (int r = 0; r < 8; ++r) {
        const int m  = r + ((lane >> 4) << 3);
        const int tg = tm * 32 + i * 16 + m;
        const int bb = tg >> 11;
        const int t  = tg & (T_ - 1);
        const int bh = bb * H_ + hh;
        const unsigned short val = f2bf(acc[i][j][r]);
        if (which == 0)      q [(size_t)(bh * T_ + t) * DH + d] = val;
        else if (which == 1) k [(size_t)(bh * T_ + t) * DH + d] = val;
        else                 vt[(size_t)(bh * DH + d) * T_ + t] = val;
      }
    }
  }
}

// ---------------- kernel 2: causal flash attention ----------------
// 128 threads = 4 waves/block; one wave per (b,h,16-query tile).
// TDM path: K (32x64) and V (64x32, from Vt) tiles DMA'd to LDS per chunk.
#define WPB 4
__global__ __launch_bounds__(128) void attn_fwd(const unsigned short* __restrict__ q,
                                                const unsigned short* __restrict__ k,
                                                const unsigned short* __restrict__ vt,
                                                unsigned short* __restrict__ ao) {
#if HAS_TDM
  __shared__ __align__(16) unsigned short ldsK[WPB][32 * DH];   // 4KB/wave
  __shared__ __align__(16) unsigned short ldsV[WPB][DH * 32];   // 4KB/wave
#endif
  __shared__ __align__(16) unsigned short lp[WPB][16 * 32];     // 1KB/wave

  const int lane  = threadIdx.x & 31;
  const int wave  = threadIdx.x >> 5;
  const int gw    = blockIdx.x * WPB + wave;   // 0 .. 8191
  const int jq    = gw & 127;                  // query tile within (b,h)
  const int bh    = gw >> 7;                   // 0 .. 63
  const int half  = lane >> 4;
  const int l15   = lane & 15;
  const int kh    = half * 8;
  const int qbase = jq * 16;

  const unsigned short* qp = q + (size_t)(bh * T_ + qbase + l15) * DH + kh;
  const v16bf aq0 = load_tile16x32(qp);
  const v16bf aq1 = load_tile16x32(qp + 32);

  v8f o0 = {}, o1 = {}, o2 = {}, o3 = {};
  float rmax[8], rsum[8];
  #pragma unroll
  for (int r = 0; r < 8; ++r) { rmax[r] = -3.0e38f; rsum[r] = 0.0f; }

  const unsigned short* kp0 = k  + (size_t)bh * T_ * DH;
  const unsigned short* vp0 = vt + (size_t)bh * DH * T_;
  unsigned short* myl = &lp[wave][0];
#if HAS_TDM
  unsigned short* myK = &ldsK[wave][0];
  unsigned short* myV = &ldsV[wave][0];
  const unsigned ldsK_off = (unsigned)(size_t)myK;  // low 32 bits = LDS offset
  const unsigned ldsV_off = (unsigned)(size_t)myV;
#endif

  const int nchunk = (qbase + 15) / 32 + 1;
  for (int c = 0; c < nchunk; ++c) {
    const int kb = c * 32;

#if HAS_TDM
    // prior chunk's DS reads must land before TDM overwrites LDS
    asm volatile("s_wait_dscnt 0" ::: "memory");
    tdm_load_2d(kp0 + (size_t)kb * DH, ldsK_off, /*tile*/DH, 32,
                /*tensor*/DH, T_, /*stride*/DH);
    tdm_load_2d(vp0 + kb, ldsV_off, /*tile*/32, DH,
                /*tensor*/T_, DH, /*stride*/T_);
    asm volatile("" ::: "memory");
    __builtin_amdgcn_s_wait_tensorcnt(0);
    asm volatile("" ::: "memory");
    v16bf b0a = load_tile16x32(myK + l15 * DH + kh);
    v16bf b0b = load_tile16x32(myK + l15 * DH + kh + 32);
    v16bf b1a = load_tile16x32(myK + (16 + l15) * DH + kh);
    v16bf b1b = load_tile16x32(myK + (16 + l15) * DH + kh + 32);
#else
    const unsigned short* kp = kp0 + (size_t)(kb + l15) * DH + kh;
    v16bf b0a = load_tile16x32(kp);
    v16bf b0b = load_tile16x32(kp + 32);
    v16bf b1a = load_tile16x32(kp + 16 * DH);
    v16bf b1b = load_tile16x32(kp + 16 * DH + 32);
#endif

    v8f s0 = {}, s1 = {};
    s0 = wmma_bf16(aq0, b0a, s0);
    s0 = wmma_bf16(aq1, b0b, s0);
    s1 = wmma_bf16(aq0, b1a, s1);
    s1 = wmma_bf16(aq1, b1b, s1);

    const bool need_mask = (kb + 31) > qbase;
    #pragma unroll
    for (int r = 0; r < 8; ++r) {
      const int m = r + half * 8;
      float v0 = s0[r] * 0.125f;   // 1/sqrt(64)
      float v1 = s1[r] * 0.125f;
      if (need_mask) {
        if (kb + l15      > qbase + m) v0 = -3.0e38f;
        if (kb + 16 + l15 > qbase + m) v1 = -3.0e38f;
      }
      s0[r] = v0; s1[r] = v1;
    }

    float corr[8];
    #pragma unroll
    for (int r = 0; r < 8; ++r) {
      float mx = fmaxf(s0[r], s1[r]);
      mx = fmaxf(mx, __shfl_xor(mx, 1, 32));
      mx = fmaxf(mx, __shfl_xor(mx, 2, 32));
      mx = fmaxf(mx, __shfl_xor(mx, 4, 32));
      mx = fmaxf(mx, __shfl_xor(mx, 8, 32));
      const float mnew = fmaxf(rmax[r], mx);
      const float cr   = __expf(rmax[r] - mnew);
      rmax[r] = mnew;
      const float p0 = __expf(s0[r] - mnew);
      const float p1 = __expf(s1[r] - mnew);
      s0[r] = p0; s1[r] = p1;
      float rs = p0 + p1;
      rs += __shfl_xor(rs, 1, 32);
      rs += __shfl_xor(rs, 2, 32);
      rs += __shfl_xor(rs, 4, 32);
      rs += __shfl_xor(rs, 8, 32);
      rsum[r] = rsum[r] * cr + rs;
      corr[r] = cr;
    }
    #pragma unroll
    for (int r = 0; r < 8; ++r) {
      o0[r] *= corr[r]; o1[r] *= corr[r];
      o2[r] *= corr[r]; o3[r] *= corr[r];
    }

    // P: C-layout f32 -> bf16 A-layout via LDS bounce
    #pragma unroll
    for (int r = 0; r < 8; ++r) {
      const int m = r + half * 8;
      myl[m * 32 + l15]      = f2bf(s0[r]);
      myl[m * 32 + 16 + l15] = f2bf(s1[r]);
    }
    asm volatile("s_wait_dscnt 0" ::: "memory");
    const v16bf pA = load_tile16x32(myl + l15 * 32 + kh);

#if HAS_TDM
    v16bf bv0 = load_tile16x32(myV + l15 * 32 + kh);
    v16bf bv1 = load_tile16x32(myV + (16 + l15) * 32 + kh);
    v16bf bv2 = load_tile16x32(myV + (32 + l15) * 32 + kh);
    v16bf bv3 = load_tile16x32(myV + (48 + l15) * 32 + kh);
#else
    const unsigned short* vp = vp0 + (size_t)l15 * T_ + kb + kh;
    v16bf bv0 = load_tile16x32(vp);
    v16bf bv1 = load_tile16x32(vp + 16 * T_);
    v16bf bv2 = load_tile16x32(vp + 32 * T_);
    v16bf bv3 = load_tile16x32(vp + 48 * T_);
#endif
    o0 = wmma_bf16(pA, bv0, o0);
    o1 = wmma_bf16(pA, bv1, o1);
    o2 = wmma_bf16(pA, bv2, o2);
    o3 = wmma_bf16(pA, bv3, o3);
  }

  const int b = bh >> 4, h = bh & (H_ - 1);
  #pragma unroll
  for (int r = 0; r < 8; ++r) {
    const float inv = 1.0f / rsum[r];
    const int m      = r + half * 8;
    const size_t row = (size_t)(b * T_ + qbase + m) * C_;
    const int cb     = h * DH + l15;
    ao[row + cb]      = f2bf(o0[r] * inv);
    ao[row + cb + 16] = f2bf(o1[r] * inv);
    ao[row + cb + 32] = f2bf(o2[r] * inv);
    ao[row + cb + 48] = f2bf(o3[r] * inv);
  }
}

// ---------------- kernel 3: output projection GEMM (32x64 per wave, f32 out) ----------------
__global__ __launch_bounds__(256) void gemm_out(const unsigned short* __restrict__ ab,
                                                const unsigned short* __restrict__ wT,
                                                float* __restrict__ out) {
  const int lane = threadIdx.x & 31;
  const int wave = threadIdx.x >> 5;
  const int w    = blockIdx.x * 8 + wave;
  const int tm   = w / 16;                 // 256 macro rows
  const int tn   = w % 16;                 // 16 macro cols
  const int l15  = lane & 15;
  const int kh   = (lane >> 4) * 8;

  const unsigned short* pa0 = ab + (size_t)(tm * 32 + l15) * C_ + kh;
  const unsigned short* pa1 = pa0 + 16 * C_;
  const unsigned short* pb  = wT + (size_t)(tn * 64 + l15) * C_ + kh;

  v8f acc[2][4] = {};
  for (int kk = 0; kk < C_; kk += 32) {
    __builtin_prefetch(pa0 + ((kk + 256) & (C_ - 1)), 0, 3);
    __builtin_prefetch(pb  + ((kk + 256) & (C_ - 1)), 0, 3);
    v16bf a0 = load_tile16x32(pa0 + kk);
    v16bf a1 = load_tile16x32(pa1 + kk);
    #pragma unroll
    for (int j = 0; j < 4; ++j) {
      v16bf b = load_tile16x32(pb + (size_t)j * 16 * C_ + kk);
      acc[0][j] = wmma_bf16(a0, b, acc[0][j]);
      acc[1][j] = wmma_bf16(a1, b, acc[1][j]);
    }
  }

  #pragma unroll
  for (int i = 0; i < 2; ++i) {
    #pragma unroll
    for (int j = 0; j < 4; ++j) {
      #pragma unroll
      for (int r = 0; r < 8; ++r) {
        const int m = r + ((lane >> 4) << 3);
        out[(size_t)(tm * 32 + i * 16 + m) * C_ + tn * 64 + j * 16 + l15] = acc[i][j][r];
      }
    }
  }
}

// ---------------- host-side launcher ----------------
extern "C" void kernel_launch(void* const* d_in, const int* in_sizes, int n_in,
                              void* d_out, int out_size, void* d_ws, size_t ws_size,
                              hipStream_t stream) {
  (void)in_sizes; (void)n_in; (void)out_size; (void)ws_size;
  const float* x    = (const float*)d_in[0];
  const float* Wqkv = (const float*)d_in[1];
  const float* Wout = (const float*)d_in[2];
  float* out = (float*)d_out;

  unsigned short* ws    = (unsigned short*)d_ws;
  unsigned short* xb    = ws;                                   // BT*C
  unsigned short* wqkvT = xb    + (size_t)BT_ * C_;             // 3C*C
  unsigned short* woutT = wqkvT + (size_t)3 * C_ * C_;          // C*C
  unsigned short* qb    = woutT + (size_t)C_ * C_;              // BT*C
  unsigned short* kb    = qb    + (size_t)BT_ * C_;             // BT*C
  unsigned short* vtb   = kb    + (size_t)BT_ * C_;             // BT*C
  unsigned short* abuf  = vtb   + (size_t)BT_ * C_;             // BT*C

  cvt_f32_bf16<<<(BT_ * C_) / 256, 256, 0, stream>>>(x, xb, BT_ * C_);
  transpose_cvt<C_, 3 * C_><<<(3 * C_ * C_) / 256, 256, 0, stream>>>(Wqkv, wqkvT);
  transpose_cvt<C_, C_><<<(C_ * C_) / 256, 256, 0, stream>>>(Wout, woutT);

  gemm_qkv<<<(256 * 48) / 8, 256, 0, stream>>>(xb, wqkvT, qb, kb, vtb);     // 1536 blocks
  attn_fwd<<<(B_ * H_ * (T_ / 16)) / WPB, WPB * 32, 0, stream>>>(qb, kb, vtb, abuf); // 2048 blocks
  gemm_out<<<(256 * 16) / 8, 256, 0, stream>>>(abuf, woutT, out);           // 512 blocks
}
